// MoE_47450798686386
// MI455X (gfx1250) — matrix operand order repaired and checked by reading, not compiled, not run on hardware
//
#include <hip/hip_runtime.h>
#include <hip/hip_bf16.h>
#include <cstdint>

// ---------------------------------------------------------------------------
// MoE (GShard top-2) for MI455X / gfx1250, wave32, bf16 WMMA pipeline with
// async global->LDS staging (ASYNCcnt) and double buffering.
//   T=2048 tokens, H=2048, I=8192, E=8, TOP_K=2
// Workspace layout (~585 MiB required in d_ws):
//   xb   : bf16 [T][H]          8 MiB
//   w1t  : bf16 [E][I][H]     256 MiB   (fc1 weights transposed, N-major)
//   w2t  : bf16 [E][H][I]     256 MiB   (fc2 weights transposed, N-major)
//   h1   : bf16 [T*K][I]       64 MiB   (segmented by expert via base[])
//   tok  : int  [E][T], gval : float [E][T], cnt/base : int[E]
// ---------------------------------------------------------------------------

#define T_TOK 2048
#define HDIM  2048
#define IDIM  8192
#define NEXP  8
#define BM    128
#define BN    128
#define BK    64                 // k-chunk staged in LDS
#define ROWQ  9                  // uint4 per LDS tile row (144 B pitch)
#define TILEQ (128 * ROWQ)       // uint4 per LDS tile (one buffer)
#define TILEB (TILEQ * 16)       // bytes per LDS tile buffer (18432)

typedef __bf16 v16bf __attribute__((ext_vector_type(16)));
typedef float  v8f   __attribute__((ext_vector_type(8)));

union FragU { v16bf v; uint4 q[2]; };

__device__ __forceinline__ int imin(int a, int b) { return a < b ? a : b; }

__device__ __forceinline__ float gelu_tanh(float x) {
    float u = 0.7978845608028654f * (x + 0.044715f * x * x * x);
    return 0.5f * x * (1.0f + tanhf(u));
}

// Async DMA: 16B per lane, global -> LDS, tracked by ASYNCcnt (CDNA5).
__device__ __forceinline__ void async_copy_b128(uint32_t lds_off, const void* gaddr) {
    asm volatile("global_load_async_to_lds_b128 %0, %1, off"
                 :: "v"(lds_off), "v"((uint64_t)(uintptr_t)gaddr)
                 : "memory");
}
__device__ __forceinline__ void wait_async0() {
    asm volatile("s_wait_asynccnt 0" ::: "memory");
}

// A fragment (16x32 bf16) from LDS tile: lane -> M=lane%16, g=lane/16.
// element i: K = (i<8) ? 8g+i : 16+8g+(i-8)  -> bytes (ks*64 + 16g) and +32
__device__ __forceinline__ v16bf lds_a_frag(const uint4* sm, int row, int ks, int g) {
    FragU f;
    int idx = row * ROWQ + ks * 4 + g;
    f.q[0] = sm[idx];
    f.q[1] = sm[idx + 2];
    return f.v;
}
// B fragment (32x16 bf16), tile stored [N][K]: lane -> N=lane%16, g=lane/16.
// element i: K = (i<8) ? 8g+i : 16+8g+(i-8)  -> bytes (ks*64 + 16g) and +32
__device__ __forceinline__ v16bf lds_b_frag(const uint4* sm, int row, int ks, int g) {
    FragU f;
    int idx = row * ROWQ + ks * 4 + g;
    f.q[0] = sm[idx];
    f.q[1] = sm[idx + 2];
    return f.v;
}

// ---------------------------------------------------------------------------
// 0) zero output + routing counters
// ---------------------------------------------------------------------------
__global__ void init_zero(float* __restrict__ out, int n, int* __restrict__ cnt) {
    int i = blockIdx.x * blockDim.x + threadIdx.x;
    if (i < NEXP) cnt[i] = 0;
    for (int idx = i; idx < n; idx += gridDim.x * blockDim.x) out[idx] = 0.0f;
}

// ---------------------------------------------------------------------------
// 1) gate: logits -> softmax -> top-2 -> per-expert token lists (1 wave/token)
// ---------------------------------------------------------------------------
__global__ __launch_bounds__(256) void gate_kernel(
    const float* __restrict__ x, const float* __restrict__ gw,
    const float* __restrict__ gb, const float* __restrict__ alpha,
    int* __restrict__ cnt, int* __restrict__ tok, float* __restrict__ gval) {
    const int lane = threadIdx.x & 31;
    const int wid  = threadIdx.x >> 5;
    const int t = blockIdx.x * 8 + wid;
    if (t >= T_TOK) return;

    float acc[NEXP];
#pragma unroll
    for (int e = 0; e < NEXP; e++) acc[e] = 0.0f;

    const float* xr = x + (size_t)t * HDIM;
    for (int h = lane; h < HDIM; h += 32) {
        float xv = xr[h];
        const float4* gp = reinterpret_cast<const float4*>(gw + (size_t)h * NEXP);
        float4 g0 = gp[0], g1 = gp[1];
        acc[0] += xv * g0.x; acc[1] += xv * g0.y;
        acc[2] += xv * g0.z; acc[3] += xv * g0.w;
        acc[4] += xv * g1.x; acc[5] += xv * g1.y;
        acc[6] += xv * g1.z; acc[7] += xv * g1.w;
    }
#pragma unroll
    for (int off = 16; off > 0; off >>= 1) {
#pragma unroll
        for (int e = 0; e < NEXP; e++) acc[e] += __shfl_xor(acc[e], off, 32);
    }

    if (lane == 0) {
        float lg[NEXP];
        float mx = -3.4e38f;
#pragma unroll
        for (int e = 0; e < NEXP; e++) { lg[e] = acc[e] + gb[e]; mx = fmaxf(mx, lg[e]); }
        float s = 0.0f, ex[NEXP];
#pragma unroll
        for (int e = 0; e < NEXP; e++) { ex[e] = expf(lg[e] - mx); s += ex[e]; }
        float inv = 1.0f / s;
        int i1 = 0; float v1 = -1.0f;
#pragma unroll
        for (int e = 0; e < NEXP; e++) {
            float sc = ex[e] * inv;
            if (sc > v1) { v1 = sc; i1 = e; }
        }
        int i2 = 0; float v2 = -1.0f;
#pragma unroll
        for (int e = 0; e < NEXP; e++) {
            float sc = ex[e] * inv;
            if (e != i1 && sc > v2) { v2 = sc; i2 = e; }
        }
        int p1 = atomicAdd(&cnt[i1], 1);
        tok[i1 * T_TOK + p1]  = t;
        gval[i1 * T_TOK + p1] = v1 * alpha[i1];
        int p2 = atomicAdd(&cnt[i2], 1);
        tok[i2 * T_TOK + p2]  = t;
        gval[i2 * T_TOK + p2] = v2 * alpha[i2];
    }
}

// ---------------------------------------------------------------------------
// 2) exclusive scan over 8 counts -> segmented h1 slot bases
// ---------------------------------------------------------------------------
__global__ void scan_kernel(const int* __restrict__ cnt, int* __restrict__ basep) {
    if (threadIdx.x == 0 && blockIdx.x == 0) {
        int s = 0;
        for (int e = 0; e < NEXP; e++) { basep[e] = s; s += cnt[e]; }
    }
}

// ---------------------------------------------------------------------------
// 3) fp32 -> bf16 activation convert
// ---------------------------------------------------------------------------
__global__ __launch_bounds__(256) void cvt_x(const float* __restrict__ x,
                                             __bf16* __restrict__ xb, int n) {
    int i = 4 * (blockIdx.x * blockDim.x + threadIdx.x);
    if (i < n) {
        float4 v = *reinterpret_cast<const float4*>(x + i);
        union { __bf16 b[4]; uint2 u; } o;
        o.b[0] = (__bf16)v.x; o.b[1] = (__bf16)v.y;
        o.b[2] = (__bf16)v.z; o.b[3] = (__bf16)v.w;
        *reinterpret_cast<uint2*>(xb + i) = o.u;
    }
}

// ---------------------------------------------------------------------------
// 4) fp32 [R][C] -> bf16 [C][R] tiled transpose-convert (per expert in grid.z)
// ---------------------------------------------------------------------------
__global__ __launch_bounds__(256) void transpose_cvt(const float* __restrict__ in,
                                                     __bf16* __restrict__ out,
                                                     int R, int C) {
    __shared__ float tile[32][33];
    const size_t eoff = (size_t)blockIdx.z * (size_t)R * (size_t)C;
    in  += eoff;
    out += eoff;
    const int r0 = blockIdx.y * 32;
    const int c0 = blockIdx.x * 32;
    const int tx = threadIdx.x & 31;
    const int ty = threadIdx.x >> 5;
#pragma unroll
    for (int i = 0; i < 4; i++)
        tile[ty + i * 8][tx] = in[(size_t)(r0 + ty + i * 8) * C + c0 + tx];
    __syncthreads();
#pragma unroll
    for (int i = 0; i < 4; i++)
        out[(size_t)(c0 + ty + i * 8) * R + r0 + tx] = (__bf16)tile[tx][ty + i * 8];
}

// ---------------------------------------------------------------------------
// 5) fc1 GEMM + GELU:  h1[slot][i] = gelu(x[tok] @ w1t[e] + b1[e])
//    128x128 block tile, BK=64 LDS chunks (async-staged, double-buffered),
//    8 waves (4m x 2n), each wave 2x4 WMMA C-tiles.
// ---------------------------------------------------------------------------
__global__ __launch_bounds__(256) void fc1_gemm(
    const __bf16* __restrict__ xb, const __bf16* __restrict__ w1t,
    const float* __restrict__ b1, const int* __restrict__ cnt,
    const int* __restrict__ basep, const int* __restrict__ tok,
    __bf16* __restrict__ h1) {
    __shared__ uint4 smA[2 * TILEQ];
    __shared__ uint4 smB[2 * TILEQ];

    const int e = blockIdx.z;
    const int rows = cnt[e];
    const int m0 = blockIdx.y * BM;
    if (m0 >= rows) return;
    const int n0 = blockIdx.x * BN;

    const int tid  = threadIdx.x;
    const int lane = tid & 31;
    const int wid  = tid >> 5;
    const int wm = wid & 3, wn = wid >> 2;
    const int ln = lane & 15, g = lane >> 4;

    // ---- staging assignment: thread -> (row 0..127, half 0/1), 4x16B segs
    const int srow = tid >> 1;
    const int half = tid & 1;
    const int trow = tok[e * T_TOK + imin(m0 + srow, rows - 1)];
    const char* gA = (const char*)(xb + (size_t)trow * HDIM) + half * 64;
    const char* gB = (const char*)(w1t + ((size_t)e * IDIM + n0 + srow) * HDIM) + half * 64;
    const uint32_t lA0 = (uint32_t)(uintptr_t)&smA[0] + srow * 144 + half * 64;
    const uint32_t lB0 = (uint32_t)(uintptr_t)&smB[0] + srow * 144 + half * 64;

    v8f acc[2][4];
#pragma unroll
    for (int mi = 0; mi < 2; mi++)
#pragma unroll
        for (int j = 0; j < 4; j++) acc[mi][j] = (v8f)0.0f;

    const int rowA0 = wm * 32 + ln;          // block-relative A rows (+16 for mi=1)
    const int rowB0 = wn * 64 + ln;          // block-relative B rows (+16j)

    const int NC = HDIM / BK;
    // prologue: stage chunk 0 into buffer 0
#pragma unroll
    for (int s = 0; s < 4; s++) {
        async_copy_b128(lA0 + s * 16, gA + s * 16);
        async_copy_b128(lB0 + s * 16, gB + s * 16);
    }
    wait_async0();
    __syncthreads();

    int buf = 0;
    for (int c = 0; c < NC; ++c) {
        if (c + 1 < NC) {
            const size_t kb = (size_t)(c + 1) * BK * 2;  // bytes
            const uint32_t lbo = (buf ^ 1) * TILEB;
#pragma unroll
            for (int s = 0; s < 4; s++) {
                async_copy_b128(lA0 + lbo + s * 16, gA + kb + s * 16);
                async_copy_b128(lB0 + lbo + s * 16, gB + kb + s * 16);
            }
        }
        const uint4* tA = smA + buf * TILEQ;
        const uint4* tB = smB + buf * TILEQ;
#pragma unroll
        for (int ks = 0; ks < 2; ++ks) {
            v16bf af0 = lds_a_frag(tA, rowA0,      ks, g);
            v16bf af1 = lds_a_frag(tA, rowA0 + 16, ks, g);
#pragma unroll
            for (int j = 0; j < 4; j++) {
                v16bf bf = lds_b_frag(tB, rowB0 + j * 16, ks, g);
                acc[0][j] = __builtin_amdgcn_wmma_f32_16x16x32_bf16(
                    false, af0, false, bf, (short)0, acc[0][j], false, false);
                acc[1][j] = __builtin_amdgcn_wmma_f32_16x16x32_bf16(
                    false, af1, false, bf, (short)0, acc[1][j], false, false);
            }
        }
        wait_async0();
        __syncthreads();
        buf ^= 1;
    }

    const int slot0 = basep[e];
    const int mw = m0 + wm * 32;
    const int nw = n0 + wn * 64;
#pragma unroll
    for (int mi = 0; mi < 2; mi++) {
#pragma unroll
        for (int r = 0; r < 8; r++) {
            int m = mw + mi * 16 + r + 8 * g;
            if (m < rows) {
                size_t rowoff = (size_t)(slot0 + m) * IDIM;
#pragma unroll
                for (int j = 0; j < 4; j++) {
                    int col = nw + j * 16 + ln;
                    float v = acc[mi][j][r] + b1[e * IDIM + col];
                    h1[rowoff + col] = (__bf16)gelu_tanh(v);
                }
            }
        }
    }
}

// ---------------------------------------------------------------------------
// 6) fc2 GEMM + weighted scatter: out[tok] += gate * (h1 @ w2t[e] + b2[e])
// ---------------------------------------------------------------------------
__global__ __launch_bounds__(256) void fc2_gemm(
    const __bf16* __restrict__ h1, const __bf16* __restrict__ w2t,
    const float* __restrict__ b2, const int* __restrict__ cnt,
    const int* __restrict__ basep, const int* __restrict__ tok,
    const float* __restrict__ gval, float* __restrict__ out) {
    __shared__ uint4 smA[2 * TILEQ];
    __shared__ uint4 smB[2 * TILEQ];

    const int e = blockIdx.z;
    const int rows = cnt[e];
    const int m0 = blockIdx.y * BM;
    if (m0 >= rows) return;
    const int n0 = blockIdx.x * BN;

    const int tid  = threadIdx.x;
    const int lane = tid & 31;
    const int wid  = tid >> 5;
    const int wm = wid & 3, wn = wid >> 2;
    const int ln = lane & 15, g = lane >> 4;

    const int slot0 = basep[e];
    const int srow = tid >> 1;
    const int half = tid & 1;
    const int aslot = slot0 + imin(m0 + srow, rows - 1);
    const char* gA = (const char*)(h1 + (size_t)aslot * IDIM) + half * 64;
    const char* gB = (const char*)(w2t + ((size_t)e * HDIM + n0 + srow) * IDIM) + half * 64;
    const uint32_t lA0 = (uint32_t)(uintptr_t)&smA[0] + srow * 144 + half * 64;
    const uint32_t lB0 = (uint32_t)(uintptr_t)&smB[0] + srow * 144 + half * 64;

    v8f acc[2][4];
#pragma unroll
    for (int mi = 0; mi < 2; mi++)
#pragma unroll
        for (int j = 0; j < 4; j++) acc[mi][j] = (v8f)0.0f;

    const int rowA0 = wm * 32 + ln;
    const int rowB0 = wn * 64 + ln;

    const int NC = IDIM / BK;
#pragma unroll
    for (int s = 0; s < 4; s++) {
        async_copy_b128(lA0 + s * 16, gA + s * 16);
        async_copy_b128(lB0 + s * 16, gB + s * 16);
    }
    wait_async0();
    __syncthreads();

    int buf = 0;
    for (int c = 0; c < NC; ++c) {
        if (c + 1 < NC) {
            const size_t kb = (size_t)(c + 1) * BK * 2;  // bytes
            const uint32_t lbo = (buf ^ 1) * TILEB;
#pragma unroll
            for (int s = 0; s < 4; s++) {
                async_copy_b128(lA0 + lbo + s * 16, gA + kb + s * 16);
                async_copy_b128(lB0 + lbo + s * 16, gB + kb + s * 16);
            }
        }
        const uint4* tA = smA + buf * TILEQ;
        const uint4* tB = smB + buf * TILEQ;
#pragma unroll
        for (int ks = 0; ks < 2; ++ks) {
            v16bf af0 = lds_a_frag(tA, rowA0,      ks, g);
            v16bf af1 = lds_a_frag(tA, rowA0 + 16, ks, g);
#pragma unroll
            for (int j = 0; j < 4; j++) {
                v16bf bf = lds_b_frag(tB, rowB0 + j * 16, ks, g);
                acc[0][j] = __builtin_amdgcn_wmma_f32_16x16x32_bf16(
                    false, af0, false, bf, (short)0, acc[0][j], false, false);
                acc[1][j] = __builtin_amdgcn_wmma_f32_16x16x32_bf16(
                    false, af1, false, bf, (short)0, acc[1][j], false, false);
            }
        }
        wait_async0();
        __syncthreads();
        buf ^= 1;
    }

    const int mw = m0 + wm * 32;
    const int nw = n0 + wn * 64;
#pragma unroll
    for (int mi = 0; mi < 2; mi++) {
#pragma unroll
        for (int r = 0; r < 8; r++) {
            int m = mw + mi * 16 + r + 8 * g;
            if (m < rows) {
                int t  = tok[e * T_TOK + m];
                float gv = gval[e * T_TOK + m];
                float* orow = out + (size_t)t * HDIM;
#pragma unroll
                for (int j = 0; j < 4; j++) {
                    int col = nw + j * 16 + ln;
                    float v = gv * (acc[mi][j][r] + b2[e * HDIM + col]);
                    atomicAdd(orow + col, v);
                }
            }
        }
    }
}

// ---------------------------------------------------------------------------
// launch
// ---------------------------------------------------------------------------
extern "C" void kernel_launch(void* const* d_in, const int* in_sizes, int n_in,
                              void* d_out, int out_size, void* d_ws, size_t ws_size,
                              hipStream_t stream) {
    const float* x     = (const float*)d_in[0];
    const float* gw    = (const float*)d_in[1];
    const float* gb    = (const float*)d_in[2];
    const float* w1    = (const float*)d_in[3];
    const float* b1    = (const float*)d_in[4];
    const float* w2    = (const float*)d_in[5];
    const float* b2    = (const float*)d_in[6];
    const float* alpha = (const float*)d_in[7];
    float* out = (float*)d_out;

    char* ws = (char*)d_ws;
    const size_t o_xb   = 0;
    const size_t o_w1t  = o_xb  + (size_t)T_TOK * HDIM * 2;            // + 8 MiB
    const size_t o_w2t  = o_w1t + (size_t)NEXP * IDIM * HDIM * 2;      // +256 MiB
    const size_t o_h1   = o_w2t + (size_t)NEXP * HDIM * IDIM * 2;      // +256 MiB
    const size_t o_tok  = o_h1  + (size_t)2 * T_TOK * IDIM * 2;        // + 64 MiB
    const size_t o_gval = o_tok  + (size_t)NEXP * T_TOK * 4;
    const size_t o_cnt  = o_gval + (size_t)NEXP * T_TOK * 4;
    const size_t o_base = o_cnt + 256;

    __bf16* xb   = (__bf16*)(ws + o_xb);
    __bf16* w1t  = (__bf16*)(ws + o_w1t);
    __bf16* w2t  = (__bf16*)(ws + o_w2t);
    __bf16* h1   = (__bf16*)(ws + o_h1);
    int*    tok  = (int*)  (ws + o_tok);
    float*  gval = (float*)(ws + o_gval);
    int*    cnt  = (int*)  (ws + o_cnt);
    int*    basep= (int*)  (ws + o_base);

    const int nOut = T_TOK * HDIM;

    init_zero<<<4096, 256, 0, stream>>>(out, nOut, cnt);
    gate_kernel<<<T_TOK / 8, 256, 0, stream>>>(x, gw, gb, alpha, cnt, tok, gval);
    scan_kernel<<<1, 32, 0, stream>>>(cnt, basep);

    cvt_x<<<(T_TOK * HDIM) / (256 * 4), 256, 0, stream>>>(x, xb, T_TOK * HDIM);
    transpose_cvt<<<dim3(IDIM / 32, HDIM / 32, NEXP), 256, 0, stream>>>(w1, w1t, HDIM, IDIM);
    transpose_cvt<<<dim3(HDIM / 32, IDIM / 32, NEXP), 256, 0, stream>>>(w2, w2t, IDIM, HDIM);

    fc1_gemm<<<dim3(IDIM / BN, T_TOK / BM, NEXP), 256, 0, stream>>>(
        xb, w1t, b1, cnt, basep, tok, h1);
    fc2_gemm<<<dim3(HDIM / BN, T_TOK / BM, NEXP), 256, 0, stream>>>(
        h1, w2t, b2, cnt, basep, tok, gval, out);
}
